// ContinuousEntityExtractor_90417651515907
// MI455X (gfx1250) — matrix operand order repaired and checked by reading, not compiled, not run on hardware
//
#include <hip/hip_runtime.h>
#include <hip/hip_bf16.h>
#include <math.h>
#include <stdint.h>

typedef __attribute__((ext_vector_type(16))) __bf16 v16bf;
typedef __attribute__((ext_vector_type(8)))  __bf16 v8bf;
typedef __attribute__((ext_vector_type(8)))  float  v8f;
typedef __attribute__((ext_vector_type(4)))  unsigned int u32x4;
typedef __attribute__((ext_vector_type(8)))  int i32x8;
typedef __attribute__((ext_vector_type(4)))  int i32x4;

#define NB      8
#define NPTS    16384
#define MTOT    (NB * NPTS)        /* 131072 rows */
#define FDIM    512
#define INDIM   514
#define K1PAD   544                /* 17 * 32 */
#define ODIM    515
#define N2PAD   640                /* 5 * 128 */
#define KSEL    1024
#define SELTOT  (NB * KSEL)        /* 8192 */

#if defined(__has_builtin)
#if __has_builtin(__builtin_amdgcn_tensor_load_to_lds)
#define HAVE_TDM 1
#if __has_include(<hip/amd_detail/amd_gfx1250_TDM.h>)
#define TDM_6ARG 1
#endif
#endif
#endif

#define WAIT_ASYNC_IMM(n) asm volatile("s_wait_asynccnt " #n ::: "memory")

// Per-lane async copy of 16 bytes global -> LDS (ASYNCcnt-tracked)
__device__ __forceinline__ void async_ld_b128(unsigned lds_off, const void* g) {
    unsigned long long ga = (unsigned long long)(uintptr_t)g;
    asm volatile("global_load_async_to_lds_b128 %0, %1, off"
                 :: "v"(lds_off), "v"(ga) : "memory");
}

#ifdef HAVE_TDM
// Issue one TDM 2D tile load: tile_d1 rows x tile_d0 bf16 elements, packed row-major into LDS.
__device__ __forceinline__ void tdm_load_tile_bf16(unsigned lds_off, const __bf16* gsrc,
                                                   unsigned tdim0, unsigned tdim1,
                                                   unsigned long long d0_stride,
                                                   unsigned tile_d0, unsigned tile_d1) {
    unsigned long long ga = (unsigned long long)(uintptr_t)gsrc;
    u32x4 g0;
    g0[0] = 1u;                                          // count=1, user descriptor, no gather
    g0[1] = lds_off;                                     // lds_addr
    g0[2] = (unsigned)(ga & 0xFFFFFFFFu);                // global_addr[31:0]
    g0[3] = (unsigned)((ga >> 32) & 0x1FFFFFFu) | (2u << 30);  // global_addr[56:32] | type=2
    i32x8 g1;
    g1[0] = 0x00010000;                                  // data_size=1 -> 2 bytes/elem
    g1[1] = (int)((tdim0 & 0xFFFFu) << 16);              // tensor_dim0[15:0]
    g1[2] = (int)((tdim0 >> 16) | ((tdim1 & 0xFFFFu) << 16));
    g1[3] = (int)((tdim1 >> 16) | (tile_d0 << 16));      // tile_dim0
    g1[4] = (int)(tile_d1 & 0xFFFFu);                    // tile_dim1 (tile_dim2=0)
    g1[5] = (int)(d0_stride & 0xFFFFFFFFull);            // tensor_dim0_stride[31:0]
    g1[6] = (int)((d0_stride >> 32) & 0xFFFFull);        // stride[47:32] (dim1_stride lo16=0)
    g1[7] = 0;
    i32x4 z4 = {0, 0, 0, 0};
#ifdef TDM_6ARG
    i32x8 z8 = {0, 0, 0, 0, 0, 0, 0, 0};
    __builtin_amdgcn_tensor_load_to_lds(g0, g1, z4, z4, z8, 0);
#else
    __builtin_amdgcn_tensor_load_to_lds(g0, g1, z4, z4, 0);
#endif
}
#endif

// ---------------------------------------------------------------- prep kernels
__global__ void prep_w1_kernel(const float* __restrict__ W1, __bf16* __restrict__ w1t) {
    int i = blockIdx.x * 256 + threadIdx.x;            // over 512*544
    if (i >= FDIM * K1PAD) return;
    int n = i / K1PAD, k = i % K1PAD;
    w1t[i] = (k < INDIM) ? (__bf16)W1[(long)k * FDIM + n] : (__bf16)0.0f;
}

__global__ void prep_w2_kernel(const float* __restrict__ W2, __bf16* __restrict__ w2t) {
    int i = blockIdx.x * 256 + threadIdx.x;            // over 640*512
    if (i >= N2PAD * FDIM) return;
    int n = i / FDIM, k = i % FDIM;
    w2t[i] = (n < ODIM) ? (__bf16)W2[(long)k * ODIM + n] : (__bf16)0.0f;
}

__global__ void zero_imp_kernel(float* __restrict__ imp) {
    int i = blockIdx.x * 256 + threadIdx.x;
    if (i < MTOT) imp[i] = 0.0f;
}

// ---------------------------------------------------------------- GEMM1: h = gelu(X @ W1 + b1)
// Block 64x128, BK=32, 8 waves (4Mx2N). Double-buffered LDS; B tiles via TDM.
__global__ __launch_bounds__(256)
void gemm1_kernel(const float* __restrict__ coords, const float* __restrict__ dens,
                  const __bf16* __restrict__ w1t, const float* __restrict__ b1,
                  __bf16* __restrict__ hout) {
    __shared__ __align__(64) __bf16 sA[2][64][32];
    __shared__ __align__(64) __bf16 sB[2][128][32];
    const int tid  = threadIdx.x;
    const int lane = tid & 31, wave = tid >> 5;
    const int wm = wave >> 1, wn = wave & 1;
    const int hi = lane >> 4, ln = lane & 15;
    const long mbase = (long)blockIdx.x * 64;
    const int  nbase = blockIdx.y * 128;

    v8f acc[4] = {};
    const int arow = tid >> 2, achk = (tid & 3) * 8;
#ifndef HAVE_TDM
    const int brow = tid >> 1, bchk = (tid & 1) * 16;
#endif
    const long grow = mbase + arow;

    float areg[8];
    // load A slice (f32, boundary-aware) for tile kt into registers
    auto loadA = [&](int kt) {
        const int kb = kt * 32;
        #pragma unroll
        for (int e = 0; e < 8; ++e) {
            int gk = kb + achk + e;
            float v;
            if (gk < 2)          v = coords[grow * 2 + gk];
            else if (gk < INDIM) v = dens[grow * (long)FDIM + (gk - 2)];
            else                 v = 0.0f;
            areg[e] = v;
        }
    };
    auto storeA = [&](int buf) {
        v8bf av;
        #pragma unroll
        for (int e = 0; e < 8; ++e) av[e] = (__bf16)areg[e];
        *(v8bf*)&sA[buf][arow][achk] = av;
    };
    auto stageB = [&](int kt, int buf) {
#ifdef HAVE_TDM
        if (wave == 0)
            tdm_load_tile_bf16((unsigned)(uintptr_t)&sB[buf][0][0],
                               w1t + (long)nbase * K1PAD + kt * 32,
                               (unsigned)(K1PAD - kt * 32), (unsigned)(FDIM - nbase),
                               (unsigned long long)K1PAD, 32u, 128u);
#else
        const long src = (long)(nbase + brow) * K1PAD + kt * 32 + bchk;
        async_ld_b128((unsigned)(uintptr_t)&sB[buf][brow][bchk], w1t + src);
#endif
    };

    loadA(0); storeA(0); stageB(0, 0);
    for (int kt = 0; kt < K1PAD / 32; ++kt) {
        const int cur = kt & 1, nxt = cur ^ 1;
        const bool more = (kt + 1 < K1PAD / 32);
        if (more) { loadA(kt + 1); stageB(kt + 1, nxt); }
#ifdef HAVE_TDM
        if (more) __builtin_amdgcn_s_wait_tensorcnt((short)1);
        else      __builtin_amdgcn_s_wait_tensorcnt((short)0);
#else
        if (more) { WAIT_ASYNC_IMM(1); } else { WAIT_ASYNC_IMM(0); }
#endif
        __syncthreads();

        const __bf16* ap = &sA[cur][wm * 16 + ln][0];
        v8bf alo = *(const v8bf*)(ap + 8 * hi);
        v8bf ahi = *(const v8bf*)(ap + 16 + 8 * hi);
        v16bf a  = __builtin_shufflevector(alo, ahi, 0,1,2,3,4,5,6,7,8,9,10,11,12,13,14,15);
        #pragma unroll
        for (int j = 0; j < 4; ++j) {
            const __bf16* bp = &sB[cur][wn * 64 + j * 16 + ln][0];
            v16bf b = __builtin_shufflevector(*(const v8bf*)(bp + 8 * hi), *(const v8bf*)(bp + 16 + 8 * hi),
                                              0,1,2,3,4,5,6,7,8,9,10,11,12,13,14,15);
            acc[j] = __builtin_amdgcn_wmma_f32_16x16x32_bf16(
                false, a, false, b, (short)0, acc[j], false, false);
        }
        if (more) storeA(nxt);   // cvt+ds_store after WMMA issue so loads hide under compute
        __syncthreads();
    }

    #pragma unroll
    for (int j = 0; j < 4; ++j) {
        const int ng   = nbase + wn * 64 + j * 16 + ln;
        const float bs = b1[ng];
        #pragma unroll
        for (int v = 0; v < 8; ++v) {
            const long mg = mbase + wm * 16 + v + 8 * hi;
            float x = acc[j][v] + bs;
            float g = 0.5f * x * (1.0f + erff(x * 0.70710678118654752f));
            hout[mg * FDIM + ng] = (__bf16)g;
        }
    }
}

// ---------------------------------------------------------------- GEMM2: importance only
__global__ __launch_bounds__(256)
void gemm2_imp_kernel(const __bf16* __restrict__ hbuf, const __bf16* __restrict__ w2t,
                      const float* __restrict__ b2, float* __restrict__ imp) {
    __shared__ __align__(64) __bf16 sA[2][64][32];
    __shared__ __align__(64) __bf16 sB[2][128][32];
    __shared__ float simp[64];
    const int tid  = threadIdx.x;
    const int lane = tid & 31, wave = tid >> 5;
    const int wm = wave >> 1, wn = wave & 1;
    const int hi = lane >> 4, ln = lane & 15;
    const long mbase = (long)blockIdx.x * 64;
    const int  nbase = blockIdx.y * 128;

    if (tid < 64) simp[tid] = 0.0f;

    v8f acc[4] = {};
    const int arow = tid >> 2, achk = (tid & 3) * 8;
#ifndef HAVE_TDM
    const int brow = tid >> 1, bchk = (tid & 1) * 16;
#endif

    auto stageA = [&](int kt, int buf) {   // pure bf16 copy -> async to LDS
        const long src = (mbase + arow) * (long)FDIM + kt * 32 + achk;
        async_ld_b128((unsigned)(uintptr_t)&sA[buf][arow][achk], hbuf + src);
    };
    auto stageB = [&](int kt, int buf) {
#ifdef HAVE_TDM
        if (wave == 0)
            tdm_load_tile_bf16((unsigned)(uintptr_t)&sB[buf][0][0],
                               w2t + (long)nbase * FDIM + kt * 32,
                               (unsigned)(FDIM - kt * 32), (unsigned)(N2PAD - nbase),
                               (unsigned long long)FDIM, 32u, 128u);
#else
        const long src = (long)(nbase + brow) * FDIM + kt * 32 + bchk;
        async_ld_b128((unsigned)(uintptr_t)&sB[buf][brow][bchk], w2t + src);
#endif
    };

    stageA(0, 0); stageB(0, 0);
    for (int kt = 0; kt < FDIM / 32; ++kt) {
        const int cur = kt & 1, nxt = cur ^ 1;
        const bool more = (kt + 1 < FDIM / 32);
        if (more) { stageA(kt + 1, nxt); stageB(kt + 1, nxt); }
#ifdef HAVE_TDM
        if (more) { WAIT_ASYNC_IMM(1); __builtin_amdgcn_s_wait_tensorcnt((short)1); }
        else      { WAIT_ASYNC_IMM(0); __builtin_amdgcn_s_wait_tensorcnt((short)0); }
#else
        if (more) { WAIT_ASYNC_IMM(3); } else { WAIT_ASYNC_IMM(0); }
#endif
        __syncthreads();

        const __bf16* ap = &sA[cur][wm * 16 + ln][0];
        v8bf alo = *(const v8bf*)(ap + 8 * hi);
        v8bf ahi = *(const v8bf*)(ap + 16 + 8 * hi);
        v16bf a  = __builtin_shufflevector(alo, ahi, 0,1,2,3,4,5,6,7,8,9,10,11,12,13,14,15);
        #pragma unroll
        for (int j = 0; j < 4; ++j) {
            const __bf16* bp = &sB[cur][wn * 64 + j * 16 + ln][0];
            v16bf b = __builtin_shufflevector(*(const v8bf*)(bp + 8 * hi), *(const v8bf*)(bp + 16 + 8 * hi),
                                              0,1,2,3,4,5,6,7,8,9,10,11,12,13,14,15);
            acc[j] = __builtin_amdgcn_wmma_f32_16x16x32_bf16(
                false, a, false, b, (short)0, acc[j], false, false);
        }
        __syncthreads();
    }

    float rs[8] = {0.f, 0.f, 0.f, 0.f, 0.f, 0.f, 0.f, 0.f};
    #pragma unroll
    for (int j = 0; j < 4; ++j) {
        const int ng = nbase + wn * 64 + j * 16 + ln;
        if (ng < ODIM) {
            const float bs = b2[ng];
            #pragma unroll
            for (int v = 0; v < 8; ++v) { float x = acc[j][v] + bs; rs[v] += x * x; }
        }
    }
    #pragma unroll
    for (int v = 0; v < 8; ++v) atomicAdd(&simp[wm * 16 + v + 8 * hi], rs[v]);
    __syncthreads();
    if (tid < 64) atomicAdd(&imp[mbase + tid], simp[tid]);
}

// ---------------------------------------------------------------- per-batch top-1024 (bitonic)
__global__ __launch_bounds__(1024)
void topk_kernel(const float* __restrict__ imp, unsigned* __restrict__ selidx) {
    extern __shared__ unsigned long long keys[];   // 16384 * 8B = 128 KB
    const int tid = threadIdx.x;
    const int b   = blockIdx.x;
    for (int i = tid; i < NPTS; i += 1024) {
        unsigned k = __float_as_uint(imp[b * NPTS + i]);   // imp >= 0: bits monotone
        keys[i] = ((unsigned long long)k << 32) | (0xFFFFFFFFu - (unsigned)i);
    }
    __syncthreads();
    for (int k = 2; k <= NPTS; k <<= 1) {
        for (int j = k >> 1; j > 0; j >>= 1) {
            for (int i = tid; i < NPTS; i += 1024) {
                int ixj = i ^ j;
                if (ixj > i) {
                    unsigned long long a = keys[i], c = keys[ixj];
                    bool desc = ((i & k) == 0);
                    if (desc ? (a < c) : (a > c)) { keys[i] = c; keys[ixj] = a; }
                }
            }
            __syncthreads();
        }
    }
    if (tid < KSEL)
        selidx[b * KSEL + tid] = 0xFFFFFFFFu - (unsigned)(keys[tid] & 0xFFFFFFFFu);
}

// ---------------------------------------------------------------- recompute selected rows & scatter
__global__ __launch_bounds__(256)
void gather_kernel(const __bf16* __restrict__ hbuf, const __bf16* __restrict__ w2t,
                   const float* __restrict__ b2, const unsigned* __restrict__ selidx,
                   float* __restrict__ out) {
    __shared__ __align__(64) __bf16 sA[2][64][32];
    __shared__ __align__(64) __bf16 sB[2][128][32];
    __shared__ unsigned srow[64];
    const int tid  = threadIdx.x;
    const int lane = tid & 31, wave = tid >> 5;
    const int wm = wave >> 1, wn = wave & 1;
    const int hi = lane >> 4, ln = lane & 15;
    const int sbase = blockIdx.x * 64;
    const int nbase = blockIdx.y * 128;

    if (tid < 64) {
        int s = sbase + tid;
        srow[tid] = (unsigned)((s >> 10) * NPTS) + selidx[s];
    }
    __syncthreads();

    v8f acc[4] = {};
    const int arow = tid >> 2, achk = (tid & 3) * 8;
#ifndef HAVE_TDM
    const int brow = tid >> 1, bchk = (tid & 1) * 16;
#endif
    const long grow = srow[arow];

    auto stageA = [&](int kt, int buf) {
        const long src = grow * FDIM + kt * 32 + achk;
        async_ld_b128((unsigned)(uintptr_t)&sA[buf][arow][achk], hbuf + src);
    };
    auto stageB = [&](int kt, int buf) {
#ifdef HAVE_TDM
        if (wave == 0)
            tdm_load_tile_bf16((unsigned)(uintptr_t)&sB[buf][0][0],
                               w2t + (long)nbase * FDIM + kt * 32,
                               (unsigned)(FDIM - kt * 32), (unsigned)(N2PAD - nbase),
                               (unsigned long long)FDIM, 32u, 128u);
#else
        const long src = (long)(nbase + brow) * FDIM + kt * 32 + bchk;
        async_ld_b128((unsigned)(uintptr_t)&sB[buf][brow][bchk], w2t + src);
#endif
    };

    stageA(0, 0); stageB(0, 0);
    for (int kt = 0; kt < FDIM / 32; ++kt) {
        const int cur = kt & 1, nxt = cur ^ 1;
        const bool more = (kt + 1 < FDIM / 32);
        if (more) { stageA(kt + 1, nxt); stageB(kt + 1, nxt); }
#ifdef HAVE_TDM
        if (more) { WAIT_ASYNC_IMM(1); __builtin_amdgcn_s_wait_tensorcnt((short)1); }
        else      { WAIT_ASYNC_IMM(0); __builtin_amdgcn_s_wait_tensorcnt((short)0); }
#else
        if (more) { WAIT_ASYNC_IMM(3); } else { WAIT_ASYNC_IMM(0); }
#endif
        __syncthreads();

        const __bf16* ap = &sA[cur][wm * 16 + ln][0];
        v8bf alo = *(const v8bf*)(ap + 8 * hi);
        v8bf ahi = *(const v8bf*)(ap + 16 + 8 * hi);
        v16bf a  = __builtin_shufflevector(alo, ahi, 0,1,2,3,4,5,6,7,8,9,10,11,12,13,14,15);
        #pragma unroll
        for (int j = 0; j < 4; ++j) {
            const __bf16* bp = &sB[cur][wn * 64 + j * 16 + ln][0];
            v16bf b = __builtin_shufflevector(*(const v8bf*)(bp + 8 * hi), *(const v8bf*)(bp + 16 + 8 * hi),
                                              0,1,2,3,4,5,6,7,8,9,10,11,12,13,14,15);
            acc[j] = __builtin_amdgcn_wmma_f32_16x16x32_bf16(
                false, a, false, b, (short)0, acc[j], false, false);
        }
        __syncthreads();
    }

    float* pos = out;
    float* st  = out + (long)SELTOT * 2;
    float* wt  = out + (long)SELTOT * 2 + (long)SELTOT * 512;
    #pragma unroll
    for (int j = 0; j < 4; ++j) {
        const int ng = nbase + wn * 64 + j * 16 + ln;
        if (ng < ODIM) {
            const float bs = b2[ng];
            #pragma unroll
            for (int v = 0; v < 8; ++v) {
                const long s = sbase + wm * 16 + v + 8 * hi;
                float val = acc[j][v] + bs;
                if (ng < 2)          pos[s * 2 + ng] = val;
                else if (ng < 514)   st[s * 512 + (ng - 2)] = val;
                else                 wt[s] = val;
            }
        }
    }
}

// ---------------------------------------------------------------- launch
extern "C" void kernel_launch(void* const* d_in, const int* in_sizes, int n_in,
                              void* d_out, int out_size, void* d_ws, size_t ws_size,
                              hipStream_t stream) {
    const float* coords = (const float*)d_in[0];
    const float* dens   = (const float*)d_in[1];
    const float* W1     = (const float*)d_in[2];
    const float* b1     = (const float*)d_in[3];
    const float* W2     = (const float*)d_in[4];
    const float* b2     = (const float*)d_in[5];
    float* out = (float*)d_out;

    char* ws = (char*)d_ws;
    __bf16*   w1t    = (__bf16*)(ws);                     // 512*544*2  = 557056
    __bf16*   w2t    = (__bf16*)(ws + 557056);            // 640*512*2  = 655360
    float*    imp    = (float*)(ws + 1212416);            // 131072*4   = 524288
    unsigned* selidx = (unsigned*)(ws + 1736704);         // 8192*4     = 32768
    __bf16*   hbuf   = (__bf16*)(ws + 1769472);           // 131072*512*2 = 134217728

    prep_w1_kernel<<<(FDIM * K1PAD + 255) / 256, 256, 0, stream>>>(W1, w1t);
    prep_w2_kernel<<<(N2PAD * FDIM + 255) / 256, 256, 0, stream>>>(W2, w2t);
    zero_imp_kernel<<<MTOT / 256, 256, 0, stream>>>(imp);

    gemm1_kernel<<<dim3(MTOT / 64, FDIM / 128), 256, 0, stream>>>(coords, dens, w1t, b1, hbuf);
    gemm2_imp_kernel<<<dim3(MTOT / 64, N2PAD / 128), 256, 0, stream>>>(hbuf, w2t, b2, imp);

    (void)hipFuncSetAttribute((const void*)topk_kernel,
                              hipFuncAttributeMaxDynamicSharedMemorySize, NPTS * 8);
    topk_kernel<<<NB, 1024, NPTS * 8, stream>>>(imp, selidx);

    gather_kernel<<<dim3(SELTOT / 64, N2PAD / 128), 256, 0, stream>>>(hbuf, w2t, b2, selidx, out);
}
